// MultiHeadAttention_50457275794016
// MI455X (gfx1250) — matrix operand order repaired and checked by reading, not compiled, not run on hardware
//
#include <hip/hip_runtime.h>
#include <hip/hip_bf16.h>

typedef __attribute__((ext_vector_type(16))) __bf16 v16bf;
typedef __attribute__((ext_vector_type(8)))  __bf16 v8bf;
typedef __attribute__((ext_vector_type(8)))  float  v8f;
typedef __attribute__((ext_vector_type(2)))  float  v2f;

#define B_  2
#define S_  2048
#define D_  512
#define H_  8
#define DK_ 64
#define M_  (B_ * S_)   // 4096

__device__ __forceinline__ v8f wmma_bf16(v16bf a, v16bf b, v8f c) {
  // D = A(16x32 bf16) x B(32x16 bf16) + C(16x16 f32)
  return __builtin_amdgcn_wmma_f32_16x16x32_bf16(false, a, false, b, (short)0, c,
                                                 false, false);
}

__device__ __forceinline__ v16bf cat8(v8bf lo, v8bf hi) {
  return __builtin_shufflevector(lo, hi, 0, 1, 2, 3, 4, 5, 6, 7,
                                          8, 9, 10, 11, 12, 13, 14, 15);
}

// ---------------------------------------------------------------------------
// GEMM: Y[M,N] = X[M,K] @ W[K,N] + bias[N]   (f32 in, bf16 math, f32 acc)
// MODE 0: write bf16 to [B,H,S,DK]   (head-split row-major)
// MODE 1: write bf16 to [B,H,DK,S]   (head-split, d-major: transposed V)
// MODE 2: write f32  to [M,N]
// Workgroup: 128 threads (4 waves), 64x64 output tile, K stepped by 32.
// ---------------------------------------------------------------------------
template <int MODE>
__global__ void __launch_bounds__(128)
gemm_bf16(const float* __restrict__ X, const float* __restrict__ W,
          const float* __restrict__ bias, void* __restrict__ outp,
          int K, int N) {
  __shared__ __align__(32) __bf16 WT[64 * 32];  // W tile transposed: [n][k]
  const int tid    = threadIdx.x;
  const int lane   = tid & 31;
  const int wid    = tid >> 5;
  const int l16    = lane & 15;
  const int hiHalf = lane >> 4;            // 0: lanes 0-15, 1: lanes 16-31
  const int aoff   = hiHalf ? 8 : 0;       // A-layout K sub-offset
  const int khalf  = hiHalf ? 16 : 0;      // B-layout K half
  const int rowoff = hiHalf ? 8 : 0;       // C-layout row offset
  const int m0 = blockIdx.x * 64;
  const int n0 = blockIdx.y * 64;

  v8f acc[4] = {v8f{}, v8f{}, v8f{}, v8f{}};
  const float* xrow = X + (size_t)(m0 + wid * 16 + l16) * K;  // A row = lane&15

  for (int k0 = 0; k0 < K; k0 += 32) {
    // stage W[k0..k0+31][n0..n0+63] -> WT[n][k] (bf16, transposed)
#pragma unroll
    for (int p = 0; p < 4; ++p) {
      const int kl = p * 8 + (tid >> 4);
      const int nl = (tid & 15) * 4;
      const float4 w4 = *(const float4*)(W + (size_t)(k0 + kl) * N + n0 + nl);
      WT[(nl + 0) * 32 + kl] = (__bf16)w4.x;
      WT[(nl + 1) * 32 + kl] = (__bf16)w4.y;
      WT[(nl + 2) * 32 + kl] = (__bf16)w4.z;
      WT[(nl + 3) * 32 + kl] = (__bf16)w4.w;
    }
    __syncthreads();

    // A operand: 16x32 tile, lane = row, K split across lane halves
    v16bf a;
#pragma unroll
    for (int e = 0; e < 8; ++e) a[e] = (__bf16)xrow[k0 + khalf + e];
#pragma unroll
    for (int e = 0; e < 8; ++e) a[8 + e] = (__bf16)xrow[k0 + khalf + 8 + e];

#pragma unroll
    for (int j = 0; j < 4; ++j) {
      v16bf bmat;
#pragma unroll
      for (int e = 0; e < 16; ++e)
        bmat[e] = WT[(16 * j + l16) * 32 + khalf + e];  // column l16 contiguous
      acc[j] = wmma_bf16(a, bmat, acc[j]);
    }
    __syncthreads();
  }

#pragma unroll
  for (int j = 0; j < 4; ++j) {
    const int n = n0 + 16 * j + l16;
    const float bn = bias[n];
#pragma unroll
    for (int r = 0; r < 8; ++r) {
      const int m = m0 + wid * 16 + r + rowoff;
      const float y = acc[j][r] + bn;
      if (MODE == 2) {
        ((float*)outp)[(size_t)m * N + n] = y;
      } else {
        const int bb = m / S_, s = m % S_, h = n / DK_, d = n % DK_;
        __bf16* o = (__bf16*)outp;
        if (MODE == 0)
          o[(((size_t)bb * H_ + h) * S_ + s) * DK_ + d] = (__bf16)y;
        else
          o[(((size_t)bb * H_ + h) * DK_ + d) * S_ + s] = (__bf16)y;
      }
    }
  }
}

// Pointwise learned rel-pos bias for one head, evaluated for TWO score
// elements at once (2-wide float vectors -> packed/dual f32 VALU ops).
__device__ __forceinline__ v2f rpe_eval2(v2f x, const float* w1,
                                         const float* b1, const float* w2,
                                         float b2) {
  v2f acc = {b2, b2};
  const v2f zero = {0.0f, 0.0f};
#pragma unroll
  for (int c = 0; c < 16; ++c) {
    const v2f w1v = {w1[c], w1[c]};
    const v2f b1v = {b1[c], b1[c]};
    const v2f w2v = {w2[c], w2[c]};
    v2f t = x * w1v + b1v;                     // fma (packed)
    t = __builtin_elementwise_max(t, zero);    // relu (packed)
    acc = acc + w2v * t;                       // fma (packed)
  }
  return acc;
}

// ---------------------------------------------------------------------------
// Flash attention: one wave handles a 16-row q tile for one (b,h).
// ---------------------------------------------------------------------------
__global__ void __launch_bounds__(32)
attn_fused(const __bf16* __restrict__ qh, const __bf16* __restrict__ kh,
           const __bf16* __restrict__ vt, const float* __restrict__ rel_pos,
           const int* __restrict__ mask, const float* __restrict__ c1w,
           const float* __restrict__ c1b, const float* __restrict__ c2w,
           const float* __restrict__ c2b, float* __restrict__ x2) {
  __shared__ __align__(32) __bf16 P[16 * 32];  // probs staging, row-major [s][t']
  const int lane   = threadIdx.x;
  const int l16    = lane & 15;
  const int hiHalf = lane >> 4;
  const int aoff   = hiHalf ? 8 : 0;
  const int khalf  = hiHalf ? 16 : 0;
  const int rowoff = hiHalf ? 8 : 0;
  const int sbase = blockIdx.x * 16;
  const int h = blockIdx.y;
  const int b = blockIdx.z;

  float w1[16], b1[16], w2[16];
#pragma unroll
  for (int c = 0; c < 16; ++c) {
    w1[c] = c1w[c];
    b1[c] = c1b[c];
    w2[c] = c2w[h * 16 + c];
  }
  const float b2 = c2b[h];

  const size_t bh = (size_t)b * H_ + h;
  const __bf16* qrow = qh + (bh * S_ + sbase + l16) * DK_;
  v16bf aQ[2];
#pragma unroll
  for (int g = 0; g < 2; ++g) {
    const v8bf lo = *(const v8bf*)(qrow + 32 * g + khalf);
    const v8bf hi = *(const v8bf*)(qrow + 32 * g + khalf + 8);
    aQ[g] = cat8(lo, hi);
  }

  v8f O[4] = {v8f{}, v8f{}, v8f{}, v8f{}};
  float mrun[8], lrun[8];
#pragma unroll
  for (int r = 0; r < 8; ++r) {
    mrun[r] = -1e30f;
    lrun[r] = 0.0f;
  }

  const float* rp0 = rel_pos + ((size_t)b * S_ + sbase) * S_;
  const int* mrow = mask + (size_t)b * S_;

  for (int t0 = 0; t0 < S_; t0 += 32) {
    // ---- scores: S = Q K^T for two 16-wide t tiles (K-dim 64 = 2 WMMAs) ----
    const __bf16* krow0 = kh + (bh * S_ + t0 + l16) * DK_;
    const __bf16* krow1 = krow0 + 16 * DK_;
    v8f sc0 = {}, sc1 = {};
    sc0 = wmma_bf16(aQ[0], *(const v16bf*)(krow0 + khalf), sc0);
    sc0 = wmma_bf16(aQ[1], *(const v16bf*)(krow0 + 32 + khalf), sc0);
    sc1 = wmma_bf16(aQ[0], *(const v16bf*)(krow1 + khalf), sc1);
    sc1 = wmma_bf16(aQ[1], *(const v16bf*)(krow1 + 32 + khalf), sc1);

    const int mk0 = mrow[t0 + l16];
    const int mk1 = mrow[t0 + 16 + l16];

    v8f corrv;
#pragma unroll
    for (int r = 0; r < 8; ++r) {
      const float* rpr = rp0 + (size_t)(r + rowoff) * S_ + t0;
      const v2f x = {rpr[l16], rpr[16 + l16]};
      const v2f bias2 = rpe_eval2(x, w1, b1, w2, b2);
      float v0 = (sc0[r] + bias2[0]) * 0.125f;
      float v1 = (sc1[r] + bias2[1]) * 0.125f;
      if (mk0 == 0) v0 = -1e9f;
      if (mk1 == 0) v1 = -1e9f;

      float rowm = fmaxf(v0, v1);
#pragma unroll
      for (int d = 1; d < 16; d <<= 1) rowm = fmaxf(rowm, __shfl_xor(rowm, d, 32));
      const float mnew = fmaxf(mrun[r], rowm);
      const float corr = __expf(mrun[r] - mnew);
      const float p0 = __expf(v0 - mnew);
      const float p1 = __expf(v1 - mnew);
      float ps = p0 + p1;
#pragma unroll
      for (int d = 1; d < 16; d <<= 1) ps += __shfl_xor(ps, d, 32);
      lrun[r] = lrun[r] * corr + ps;
      mrun[r] = mnew;
      corrv[r] = corr;
      // C-layout -> LDS (row-major): scattered b16 stores
      P[(r + rowoff) * 32 + l16] = (__bf16)p0;
      P[(r + rowoff) * 32 + 16 + l16] = (__bf16)p1;
    }

    // vectorized flash rescale (packed f32 muls)
#pragma unroll
    for (int j = 0; j < 4; ++j) O[j] = O[j] * corrv;

    // ---- re-read P in A layout (single wave: LDS is in-order) ----
    v16bf aP;
#pragma unroll
    for (int e = 0; e < 8; ++e) aP[e] = P[l16 * 32 + khalf + e];
#pragma unroll
    for (int e = 0; e < 8; ++e) aP[8 + e] = P[l16 * 32 + khalf + 8 + e];

    // ---- O += P @ V  (V pre-transposed: column d contiguous over t) ----
#pragma unroll
    for (int j = 0; j < 4; ++j) {
      const __bf16* vrow = vt + (bh * DK_ + 16 * j + l16) * S_ + t0 + khalf;
      O[j] = wmma_bf16(aP, *(const v16bf*)vrow, O[j]);
    }
  }

  // ---- normalize (vectorized), write f32 [B,S,D] for output projection ----
  v8f invv;
#pragma unroll
  for (int r = 0; r < 8; ++r) invv[r] = 1.0f / lrun[r];
#pragma unroll
  for (int j = 0; j < 4; ++j) O[j] = O[j] * invv;
#pragma unroll
  for (int r = 0; r < 8; ++r) {
    const int srow = sbase + r + rowoff;
    float* orow = x2 + ((size_t)b * S_ + srow) * D_ + h * DK_;
#pragma unroll
    for (int j = 0; j < 4; ++j) orow[16 * j + l16] = O[j][r];
  }
}

extern "C" void kernel_launch(void* const* d_in, const int* in_sizes, int n_in,
                              void* d_out, int out_size, void* d_ws,
                              size_t ws_size, hipStream_t stream) {
  (void)in_sizes; (void)n_in; (void)out_size; (void)ws_size;
  const float* Qin = (const float*)d_in[0];
  const float* Kin = (const float*)d_in[1];
  const float* Vin = (const float*)d_in[2];
  const float* rel = (const float*)d_in[3];
  const int*   msk = (const int*)d_in[4];
  const float* Wq = (const float*)d_in[5];
  const float* bq = (const float*)d_in[6];
  const float* Wk = (const float*)d_in[7];
  const float* bk = (const float*)d_in[8];
  const float* Wv = (const float*)d_in[9];
  const float* bv = (const float*)d_in[10];
  const float* Wo = (const float*)d_in[11];
  const float* bo = (const float*)d_in[12];
  const float* c1w = (const float*)d_in[13];
  const float* c1b = (const float*)d_in[14];
  const float* c2w = (const float*)d_in[15];
  const float* c2b = (const float*)d_in[16];

  char* ws = (char*)d_ws;
  __bf16* qh = (__bf16*)(ws + (size_t)0);          // 4 MB  bf16 [B,H,S,DK]
  __bf16* kh = (__bf16*)(ws + ((size_t)4 << 20));  // 4 MB  bf16 [B,H,S,DK]
  __bf16* vt = (__bf16*)(ws + ((size_t)8 << 20));  // 4 MB  bf16 [B,H,DK,S]
  float*  x2 = (float*)(ws + ((size_t)12 << 20));  // 8 MB  f32  [B,S,D]

  dim3 gg(M_ / 64, D_ / 64);
  gemm_bf16<0><<<gg, 128, 0, stream>>>(Qin, Wq, bq, (void*)qh, D_, D_);
  gemm_bf16<0><<<gg, 128, 0, stream>>>(Kin, Wk, bk, (void*)kh, D_, D_);
  gemm_bf16<1><<<gg, 128, 0, stream>>>(Vin, Wv, bv, (void*)vt, D_, D_);
  attn_fused<<<dim3(S_ / 16, H_, B_), 32, 0, stream>>>(qh, kh, vt, rel, msk,
                                                       c1w, c1b, c2w, c2b, x2);
  gemm_bf16<2><<<gg, 128, 0, stream>>>(x2, Wo, bo, d_out, D_, D_);
}
